// VSSBlock_23519240913611
// MI455X (gfx1250) — matrix operand order repaired and checked by reading, not compiled, not run on hardware
//
#include <hip/hip_runtime.h>

typedef _Float16 v16h __attribute__((ext_vector_type(16)));
typedef _Float16 v8h  __attribute__((ext_vector_type(8)));
typedef float    v8f  __attribute__((ext_vector_type(8)));
typedef int      i32x4 __attribute__((vector_size(16)));

constexpr int L_ = 3136, H_ = 56, W_ = 56, C_ = 96, DI_ = 192;
constexpr int K_ = 4, N_ = 16, DR_ = 6, MH_ = 384;

__device__ __forceinline__ float siluf(float v) { return v / (1.f + expf(-v)); }
__device__ __forceinline__ float geluf(float v) {
    const float c = 0.7978845608028654f;
    return 0.5f * v * (1.f + tanhf(c * (v + 0.044715f * v * v * v)));
}

// ---------------- LayerNorm (optional elementwise multiply), f32 -> f16 -----
__global__ void k_ln_h(const float* __restrict__ x, const float* __restrict__ g,
                       const float* __restrict__ b, const float* __restrict__ mul,
                       _Float16* __restrict__ out, int cols) {
    __shared__ float red[64];
    int row = blockIdx.x, tid = threadIdx.x;
    const float* xr = x + (long)row * cols;
    float s = 0.f;
    for (int c = tid; c < cols; c += 64) s += xr[c];
    red[tid] = s; __syncthreads();
    for (int st = 32; st > 0; st >>= 1) { if (tid < st) red[tid] += red[tid + st]; __syncthreads(); }
    float mean = red[0] / cols; __syncthreads();
    float v = 0.f;
    for (int c = tid; c < cols; c += 64) { float d = xr[c] - mean; v += d * d; }
    red[tid] = v; __syncthreads();
    for (int st = 32; st > 0; st >>= 1) { if (tid < st) red[tid] += red[tid + st]; __syncthreads(); }
    float rstd = rsqrtf(red[0] / cols + 1e-5f);
    for (int c = tid; c < cols; c += 64) {
        float val = (xr[c] - mean) * rstd * g[c] + b[c];
        if (mul) val *= mul[(long)row * cols + c];
        out[(long)row * cols + c] = (_Float16)val;
    }
}

// ---------------- Weight convert: f32 (Kd x Nd) -> f16 transposed (Nd x Kd) -
__global__ void k_w_t(const float* __restrict__ W, _Float16* __restrict__ Wt,
                      int Kd, int Nd) {
    int i = blockIdx.x * 256 + threadIdx.x;
    if (i >= Kd * Nd) return;
    int n = i % Nd, k = i / Nd;
    Wt[(long)n * Kd + k] = (_Float16)W[i];
}

// x_proj_w (K,38,192) f32 -> (K,48,192) f16, zero-padded rows 38..47
__global__ void k_xpw(const float* __restrict__ W, _Float16* __restrict__ Wh) {
    int i = blockIdx.x * 256 + threadIdx.x;
    if (i >= K_ * 48 * DI_) return;
    int d = i % DI_; int rc = (i / DI_) % 48; int k = i / (DI_ * 48);
    float v = (rc < (DR_ + 2 * N_)) ? W[((long)k * (DR_ + 2 * N_) + rc) * DI_ + d] : 0.f;
    Wh[i] = (_Float16)v;
}

// gfx1250 async global->LDS 16-byte copy (ASYNCcnt-tracked), with fallback.
__device__ __forceinline__ void async_ld_b128(const _Float16* g, _Float16* l) {
#if __has_builtin(__builtin_amdgcn_global_load_async_to_lds_b128)
    __builtin_amdgcn_global_load_async_to_lds_b128(
        (i32x4 __attribute__((address_space(1))) *)g,
        (i32x4 __attribute__((address_space(3))) *)l, 0, 0);
#else
    *(v8h*)l = *(const v8h*)g;
#endif
}

__device__ __forceinline__ void async_wait0() {
#if __has_builtin(__builtin_amdgcn_global_load_async_to_lds_b128)
#if __has_builtin(__builtin_amdgcn_s_wait_asynccnt)
    __builtin_amdgcn_s_wait_asynccnt(0);
#else
    asm volatile("s_wait_asynccnt 0x0" ::: "memory");
#endif
#endif
}

// ---------------- WMMA GEMM: C[M,N] = A[M,K] * Bt[N,K]^T (+bias, act, res) --
// 64 threads (2 waves). B panel (32 cols x Kd) is staged into LDS once per
// block via the gfx1250 async-to-LDS path, then each wave's B fragment is
// served from ds_load. Each wave computes MT M-tiles against one N-tile so
// every B fragment feeds MT v_wmma issues. Batched over blockIdx.z.
template <int MT>
__global__ void k_gemm(const _Float16* __restrict__ A, const _Float16* __restrict__ Bt,
                       float* __restrict__ C, _Float16* __restrict__ Ch,
                       const float* __restrict__ bias, const float* __restrict__ res,
                       int M, int Nn, int Kd, int act,
                       long sA, long sB, long sC) {
    __shared__ __attribute__((aligned(16))) _Float16 bsh[32 * 384];
    A  += (long)blockIdx.z * sA;
    Bt += (long)blockIdx.z * sB;
    long cOff = (long)blockIdx.z * sC;
    int tid  = threadIdx.x;
    int lane = tid & 31;
    int wv   = tid >> 5;
    int r    = lane & 15;
    int hf   = lane >> 4;
    int row0 = blockIdx.x * (16 * MT);
    int col0 = blockIdx.y * 32;
    int col  = col0 + wv * 16 + r;

    // Stage B panel -> LDS (32*Kd halves; chunk count is a multiple of 64,
    // so the loop is convergent and EXEC stays all-ones for the WMMAs below).
    {
        const _Float16* bbase = Bt + (long)col0 * Kd;
        int chunks = (32 * Kd) >> 3;   // 16-byte chunks
        for (int c = tid; c < chunks; c += 64)
            async_ld_b128(bbase + (long)c * 8, bsh + c * 8);
        async_wait0();
        __syncthreads();
    }

    // A 16x32 f16 layout: lanes 0-15 -> K 0..15 ; lanes 16-31 -> K 16..31
    const _Float16* ap[MT];
#pragma unroll
    for (int mt = 0; mt < MT; mt++)
        ap[mt] = A + (long)(row0 + mt * 16 + r) * Kd + hf * 16;
    // B 32x16 f16 layout: lanes 0-15 -> K 0..15 ; lanes 16-31 -> K 16..31
    const _Float16* bl = bsh + (wv * 16 + r) * Kd + hf * 16;

    v8f acc[MT];
#pragma unroll
    for (int mt = 0; mt < MT; mt++)
        acc[mt] = (v8f){0.f, 0.f, 0.f, 0.f, 0.f, 0.f, 0.f, 0.f};

    for (int kk = 0; kk < Kd; kk += 32) {
        v16h bv = *(const v16h*)(bl + kk);           // ds_load from staged panel
#pragma unroll
        for (int mt = 0; mt < MT; mt++) {
            if (kk + 32 < Kd)
                __builtin_prefetch((const void*)(ap[mt] + kk + 32), 0, 0);
            v8h alo = *(const v8h*)(ap[mt] + kk);
            v8h ahi = *(const v8h*)(ap[mt] + kk + 8);
            v16h av;
#pragma unroll
            for (int i = 0; i < 8; i++) { av[i] = alo[i]; av[i + 8] = ahi[i]; }
            acc[mt] = __builtin_amdgcn_wmma_f32_16x16x32_f16(false, av, false, bv,
                                                             (short)0, acc[mt], false, false);
        }
    }
    float bc = bias ? bias[col] : 0.f;
#pragma unroll
    for (int mt = 0; mt < MT; mt++) {
#pragma unroll
        for (int v = 0; v < 8; v++) {
            int row = row0 + mt * 16 + v + 8 * hf;   // C/D: VGPR v -> M = v + 8*hf, N = r
            float val = acc[mt][v] + bc;
            if (act == 1) val = geluf(val);
            if (res) val += res[(long)row * Nn + col];
            C[cOff + (long)row * Nn + col] = val;
            if (Ch) Ch[(long)row * Nn + col] = (_Float16)val;
        }
    }
}

// ---------------- Pointwise stages ------------------------------------------
__global__ void k_silu_z(const float* __restrict__ xz, float* __restrict__ z) {
    int i = blockIdx.x * 256 + threadIdx.x;
    if (i >= L_ * DI_) return;
    int d = i % DI_, l = i / DI_;
    z[i] = siluf(xz[(long)l * (2 * DI_) + DI_ + d]);
}

__global__ void k_dwconv(const float* __restrict__ xz, const float* __restrict__ cw,
                         const float* __restrict__ cb, float* __restrict__ xc) {
    int i = blockIdx.x * 256 + threadIdx.x;
    if (i >= L_ * DI_) return;
    int d = i % DI_, l = i / DI_;
    int h = l / W_, w = l % W_;
    float acc = cb[d];
    for (int dy = 0; dy < 3; dy++) {
        int hh = h + dy - 1; if (hh < 0 || hh >= H_) continue;
        for (int dx = 0; dx < 3; dx++) {
            int ww = w + dx - 1; if (ww < 0 || ww >= W_) continue;
            acc += cw[(dy * 3 + dx) * DI_ + d] * xz[(long)(hh * W_ + ww) * (2 * DI_) + d];
        }
    }
    xc[i] = siluf(acc);
}

__global__ void k_build_xs(const float* __restrict__ xc, _Float16* __restrict__ xs) {
    int i = blockIdx.x * 256 + threadIdx.x;
    if (i >= L_ * DI_) return;
    int d = i % DI_, l = i / DI_;
    int h = l / W_, w = l % W_;
    int lt = w * H_ + h;
    _Float16 v = (_Float16)xc[i];
    long S = (long)L_ * DI_;
    xs[0 * S + (long)l * DI_ + d]             = v;
    xs[1 * S + (long)lt * DI_ + d]            = v;
    xs[2 * S + (long)(L_ - 1 - l) * DI_ + d]  = v;
    xs[3 * S + (long)(L_ - 1 - lt) * DI_ + d] = v;
}

__global__ void k_dt(const float* __restrict__ xdbl, const float* __restrict__ dtw,
                     const float* __restrict__ dtb, float* __restrict__ dts) {
    int i = blockIdx.x * 256 + threadIdx.x;
    if (i >= K_ * DI_ * L_) return;
    int l = i % L_; int kd = i / L_;
    int k = kd / DI_;
    const float* xb = xdbl + (long)k * 48 * L_;
    float s = dtb[kd];
#pragma unroll
    for (int rr = 0; rr < DR_; rr++)
        s += dtw[(long)kd * DR_ + rr] * xb[(long)rr * L_ + l];
    dts[i] = (s > 20.f) ? s : log1pf(expf(s));
}

__global__ void k_scan(const float* __restrict__ dts, const float* __restrict__ xdbl,
                       const _Float16* __restrict__ xs, const float* __restrict__ A_logs,
                       const float* __restrict__ Ds, float* __restrict__ ys) {
    int kd = blockIdx.x * 64 + threadIdx.x;
    if (kd >= K_ * DI_) return;
    int k = kd / DI_, d = kd % DI_;
    float Amat[N_];
#pragma unroll
    for (int n = 0; n < N_; n++) Amat[n] = -expf(A_logs[(long)kd * N_ + n]);
    float Dv = Ds[kd];
    float hst[N_];
#pragma unroll
    for (int n = 0; n < N_; n++) hst[n] = 0.f;
    const float* dtp = dts + (long)kd * L_;
    const float* Bp = xdbl + ((long)k * 48 + DR_) * L_;
    const float* Cp = xdbl + ((long)k * 48 + DR_ + N_) * L_;
    const _Float16* xp = xs + (long)k * L_ * DI_ + d;
    for (int l = 0; l < L_; l++) {
        float dt = dtp[l];
        float xv = (float)xp[(long)l * DI_];
        float y = 0.f;
#pragma unroll
        for (int n = 0; n < N_; n++) {
            float a = expf(dt * Amat[n]);
            hst[n] = a * hst[n] + dt * Bp[(long)n * L_ + l] * xv;
            y += hst[n] * Cp[(long)n * L_ + l];
        }
        ys[(long)kd * L_ + l] = y + Dv * xv;
    }
}

__global__ void k_combine(const float* __restrict__ ys, float* __restrict__ yc) {
    int i = blockIdx.x * 256 + threadIdx.x;
    if (i >= L_ * DI_) return;
    int d = i % DI_, l = i / DI_;
    int h = l / W_, w = l % W_;
    int lt = w * H_ + h;
    long S = (long)DI_ * L_;
    float v = ys[0 * S + (long)d * L_ + l]
            + ys[2 * S + (long)d * L_ + (L_ - 1 - l)]
            + ys[1 * S + (long)d * L_ + lt]
            + ys[3 * S + (long)d * L_ + (L_ - 1 - lt)];
    yc[(long)l * DI_ + d] = v;
}

// ---------------------------------------------------------------------------
extern "C" void kernel_launch(void* const* d_in, const int* in_sizes, int n_in,
                              void* d_out, int out_size, void* d_ws, size_t ws_size,
                              hipStream_t stream) {
    (void)in_sizes; (void)n_in; (void)out_size; (void)ws_size;
    const float* x         = (const float*)d_in[0];
    const float* norm_g    = (const float*)d_in[1];
    const float* norm_b    = (const float*)d_in[2];
    const float* in_proj_w = (const float*)d_in[3];
    const float* conv_w    = (const float*)d_in[4];
    const float* conv_b    = (const float*)d_in[5];
    const float* x_proj_w  = (const float*)d_in[6];
    const float* dt_w      = (const float*)d_in[7];
    const float* dt_b      = (const float*)d_in[8];
    const float* A_logs    = (const float*)d_in[9];
    const float* Ds        = (const float*)d_in[10];
    const float* on_g      = (const float*)d_in[11];
    const float* on_b      = (const float*)d_in[12];
    const float* out_proj_w= (const float*)d_in[13];
    const float* n2_g      = (const float*)d_in[14];
    const float* n2_b      = (const float*)d_in[15];
    const float* fc1_w     = (const float*)d_in[16];
    const float* fc1_b     = (const float*)d_in[17];
    const float* fc2_w     = (const float*)d_in[18];
    const float* fc2_b     = (const float*)d_in[19];
    float* out = (float*)d_out;

    char* ws = (char*)d_ws;
    size_t o = 0;
    auto alloc = [&](size_t bytes) -> char* {
        char* p = ws + o; o = (o + bytes + 255) & ~(size_t)255; return p;
    };
    _Float16* XN_H   = (_Float16*)alloc((size_t)L_ * C_ * 2);
    _Float16* WT_IN  = (_Float16*)alloc((size_t)2 * DI_ * C_ * 2);
    float*    XZ     = (float*)   alloc((size_t)L_ * 2 * DI_ * 4);
    float*    ZS     = (float*)   alloc((size_t)L_ * DI_ * 4);
    float*    XC     = (float*)   alloc((size_t)L_ * DI_ * 4);
    _Float16* XS     = (_Float16*)alloc((size_t)4 * L_ * DI_ * 2);
    _Float16* XPW    = (_Float16*)alloc((size_t)K_ * 48 * DI_ * 2);
    float*    XDBL   = (float*)   alloc((size_t)K_ * 48 * L_ * 4);
    float*    DTS    = (float*)   alloc((size_t)K_ * DI_ * L_ * 4);
    float*    YS     = (float*)   alloc((size_t)K_ * DI_ * L_ * 4);
    float*    YC     = (float*)   alloc((size_t)L_ * DI_ * 4);
    _Float16* YLN    = (_Float16*)alloc((size_t)L_ * DI_ * 2);
    _Float16* WT_OUT = (_Float16*)alloc((size_t)C_ * DI_ * 2);
    float*    X1     = (float*)   alloc((size_t)L_ * C_ * 4);
    _Float16* H2     = (_Float16*)alloc((size_t)L_ * C_ * 2);
    _Float16* WT_F1  = (_Float16*)alloc((size_t)MH_ * C_ * 2);
    float*    HM     = (float*)   alloc((size_t)L_ * MH_ * 4);
    _Float16* HMH    = (_Float16*)alloc((size_t)L_ * MH_ * 2);
    _Float16* WT_F2  = (_Float16*)alloc((size_t)C_ * MH_ * 2);

    auto cdiv = [](int a, int b) { return (a + b - 1) / b; };

    // Weight conversions (f32 -> f16 transposed) + x_proj pad
    k_w_t<<<cdiv(C_ * 2 * DI_, 256), 256, 0, stream>>>(in_proj_w, WT_IN, C_, 2 * DI_);
    k_w_t<<<cdiv(DI_ * C_, 256), 256, 0, stream>>>(out_proj_w, WT_OUT, DI_, C_);
    k_w_t<<<cdiv(C_ * MH_, 256), 256, 0, stream>>>(fc1_w, WT_F1, C_, MH_);
    k_w_t<<<cdiv(MH_ * C_, 256), 256, 0, stream>>>(fc2_w, WT_F2, MH_, C_);
    k_xpw<<<cdiv(K_ * 48 * DI_, 256), 256, 0, stream>>>(x_proj_w, XPW);

    // LN1 -> in_proj (WMMA): xz = ln(x) @ in_proj_w   [3136 x 384]
    k_ln_h<<<dim3(L_), 64, 0, stream>>>(x, norm_g, norm_b, nullptr, XN_H, C_);
    k_gemm<2><<<dim3(L_ / 32, (2 * DI_) / 32, 1), 64, 0, stream>>>(
        XN_H, WT_IN, XZ, nullptr, nullptr, nullptr, L_, 2 * DI_, C_, 0, 0, 0, 0);

    // z = silu(gate); xx = silu(dwconv3x3(xx)); build 4 direction variants
    k_silu_z<<<cdiv(L_ * DI_, 256), 256, 0, stream>>>(XZ, ZS);
    k_dwconv<<<cdiv(L_ * DI_, 256), 256, 0, stream>>>(XZ, conv_w, conv_b, XC);
    k_build_xs<<<cdiv(L_ * DI_, 256), 256, 0, stream>>>(XC, XS);

    // x_proj (WMMA, batched over K=4): xdbl[k] = Wpad[k](48x192) @ xs[k](192xL)
    k_gemm<1><<<dim3(48 / 16, L_ / 32, K_), 64, 0, stream>>>(
        XPW, XS, XDBL, nullptr, nullptr, nullptr, 48, L_, DI_, 0,
        (long)48 * DI_, (long)L_ * DI_, (long)48 * L_);

    // dt projection + softplus, then the sequential selective scan
    k_dt<<<cdiv(K_ * DI_ * L_, 256), 256, 0, stream>>>(XDBL, dt_w, dt_b, DTS);
    k_scan<<<cdiv(K_ * DI_, 64), 64, 0, stream>>>(DTS, XDBL, XS, A_logs, Ds, YS);
    k_combine<<<cdiv(L_ * DI_, 256), 256, 0, stream>>>(YS, YC);

    // out_norm * z -> out_proj (WMMA) + residual x  => X1 [3136 x 96]
    k_ln_h<<<dim3(L_), 64, 0, stream>>>(YC, on_g, on_b, ZS, YLN, DI_);
    k_gemm<2><<<dim3(L_ / 32, C_ / 32, 1), 64, 0, stream>>>(
        YLN, WT_OUT, X1, nullptr, nullptr, x, L_, C_, DI_, 0, 0, 0, 0);

    // MLP: LN2 -> fc1+gelu (WMMA) -> fc2 (WMMA) + residual X1 => out
    k_ln_h<<<dim3(L_), 64, 0, stream>>>(X1, n2_g, n2_b, nullptr, H2, C_);
    k_gemm<2><<<dim3(L_ / 32, MH_ / 32, 1), 64, 0, stream>>>(
        H2, WT_F1, HM, HMH, fc1_b, nullptr, L_, MH_, C_, 1, 0, 0, 0);
    k_gemm<2><<<dim3(L_ / 32, C_ / 32, 1), 64, 0, stream>>>(
        HMH, WT_F2, out, nullptr, fc2_b, X1, L_, C_, MH_, 0, 0, 0, 0);
}
